// Node_12902081757248
// MI455X (gfx1250) — matrix operand order repaired and checked by reading, not compiled, not run on hardware
//
#include <hip/hip_runtime.h>
#include <hip/hip_bf16.h>
#include <stdint.h>

// ---------------------------------------------------------------------------
// Types for CDNA5 WMMA (wave32): v16bf A/B operands, v8f accumulator.
// ---------------------------------------------------------------------------
typedef __attribute__((ext_vector_type(16))) __bf16 v16bf;
typedef __attribute__((ext_vector_type(8)))  float  v8f;
typedef __attribute__((ext_vector_type(4)))  unsigned int u32x4;

struct bf32pack { u32x4 lo; u32x4 hi; };   // 32 bytes == v16bf

__device__ __forceinline__ v16bf load_tile16(const unsigned short* p0,
                                             const unsigned short* p1) {
  // Two 16-byte chunks (8 bf16 each) -> one v16bf operand.
  bf32pack t;
  t.lo = *reinterpret_cast<const u32x4*>(p0);
  t.hi = *reinterpret_cast<const u32x4*>(p1);
  return __builtin_bit_cast(v16bf, t);
}

__device__ __forceinline__ v8f wmma_bf16(v16bf a, v16bf b, v8f c) {
  // D = A(16x32 bf16) x B(32x16 bf16) + C(16x16 f32)
  return __builtin_amdgcn_wmma_f32_16x16x32_bf16(
      /*neg_a=*/false, a, /*neg_b=*/false, b,
      /*c_mod=*/(short)0, c, /*reuse_a=*/false, /*reuse_b=*/false);
}

__device__ __forceinline__ unsigned short f32_to_bf16_rne(float f) {
  unsigned u = __float_as_uint(f);
  unsigned r = u + 0x7FFFu + ((u >> 16) & 1u);
  return (unsigned short)(r >> 16);
}
__device__ __forceinline__ float bf16_to_f32(unsigned short h) {
  return __uint_as_float(((unsigned)h) << 16);
}

// ---------------------------------------------------------------------------
// Preprocess: f32 rows -> bf16 hi/lo split (+ optional row sum-of-squares).
// One block per row, 256 threads.
// ---------------------------------------------------------------------------
__global__ __launch_bounds__(256) void split_rows_kernel(
    const float* __restrict__ src, unsigned short* __restrict__ hi,
    unsigned short* __restrict__ lo, float* __restrict__ sumsq, int cols) {
  const int row = blockIdx.x;
  const float* s = src + (size_t)row * cols;
  unsigned short* h = hi + (size_t)row * cols;
  unsigned short* l = lo + (size_t)row * cols;
  float acc = 0.0f;
  for (int c = threadIdx.x; c < cols; c += blockDim.x) {
    float f = s[c];
    unsigned short hb = f32_to_bf16_rne(f);
    float hf = bf16_to_f32(hb);
    h[c] = hb;
    l[c] = f32_to_bf16_rne(f - hf);
    acc += f * f;
  }
  if (sumsq != nullptr) {
    __shared__ float red[256];
    red[threadIdx.x] = acc;
    __syncthreads();
    for (int s2 = 128; s2 > 0; s2 >>= 1) {
      if (threadIdx.x < s2) red[threadIdx.x] += red[threadIdx.x + s2];
      __syncthreads();
    }
    if (threadIdx.x == 0) sumsq[row] = red[0];
  }
}

// Split W [D,K] into bf16 hi/lo stored TRANSPOSED as [K,D] so that the
// B operand of the fallback GEMM is contiguous along the reduction axis.
__global__ __launch_bounds__(256) void split_w_transpose_kernel(
    const float* __restrict__ W, unsigned short* __restrict__ wh,
    unsigned short* __restrict__ wl, int Dd, int Kk) {
  int i = blockIdx.x * blockDim.x + threadIdx.x;
  if (i >= Dd * Kk) return;
  int d = i / Kk, c = i - d * Kk;
  float f = W[i];
  unsigned short hb = f32_to_bf16_rne(f);
  wh[(size_t)c * Dd + d] = hb;
  wl[(size_t)c * Dd + d] = f32_to_bf16_rne(f - bf16_to_f32(hb));
}

// ---------------------------------------------------------------------------
// Kernel A: fused distance GEMM + per-row argmin.
//   - 32 x-rows per workgroup (grid.x = N/32), 8 waves: wave = (rowTile 0..1,
//     colTile 0..3) -> 32x64 tile of scores per M-chunk.
//   - Inner: 16 K-steps x 3 split-bf16 WMMAs on 3 independent accumulators
//     (hi*hi, hi*lo, lo*hi) so the XDL chains don't serialize.
//   - Running argmin metric is c2[col] - 2*x.c (||x||^2 and sqrt are
//     monotone per row -> applied once at the end). No TRANS ops in the loop.
// ---------------------------------------------------------------------------
__global__ __launch_bounds__(256) void dist_argmin_kernel(
    const unsigned short* __restrict__ xh, const unsigned short* __restrict__ xl,
    const unsigned short* __restrict__ ch, const unsigned short* __restrict__ cl,
    const float* __restrict__ x2, const float* __restrict__ c2,
    float* __restrict__ minloss, int* __restrict__ minidx, int M) {
  constexpr int D = 512;
  const int tid = threadIdx.x;
  const int lane = tid & 31;
  const int wave = tid >> 5;
  const int rowTile = wave >> 2;        // 0..1
  const int colTile = wave & 3;         // 0..3
  const int rowBase = blockIdx.x * 32;
  const int l = lane & 15;
  const int koffA = (lane >> 4) * 8;    // A: lanes 0-15 K{0-7,16-23}; 16-31 K{8-15,24-31}
  const int koffB = (lane >> 4) * 8;    // B: lanes 0-15 K{0-7,16-23}; 16-31 K{8-15,24-31}

  const int aRow = rowBase + rowTile * 16 + l;
  const unsigned short* xha = xh + (size_t)aRow * D;
  const unsigned short* xla = xl + (size_t)aRow * D;

  float rmin[8];                        // running metric: c2 - 2*x.c
  int ridx[8];
#pragma unroll
  for (int v = 0; v < 8; ++v) { rmin[v] = 3.4e38f; ridx[v] = 0; }

  const v8f zero = {0.f, 0.f, 0.f, 0.f, 0.f, 0.f, 0.f, 0.f};

  for (int mBase = 0; mBase < M; mBase += 64) {
    const int col = mBase + colTile * 16 + l;     // this lane's B column (cache row)
    const unsigned short* chb = ch + (size_t)col * D;
    const unsigned short* clb = cl + (size_t)col * D;
    // Prefetch next chunk's B rows (gfx1250 global_prefetch_b8; L2-resident).
    __builtin_prefetch(chb + (size_t)64 * D, 0, 1);

    v8f acc_hh = zero, acc_hl = zero, acc_lh = zero;
#pragma unroll
    for (int ks = 0; ks < D; ks += 32) {
      const int a0 = ks + koffA, a1 = ks + koffA + 16;
      const int b0 = ks + koffB, b1 = ks + koffB + 16;
      v16bf ah = load_tile16(xha + a0, xha + a1);
      v16bf al = load_tile16(xla + a0, xla + a1);
      v16bf bh = load_tile16(chb + b0, chb + b1);
      v16bf bl = load_tile16(clb + b0, clb + b1);
      acc_hh = wmma_bf16(ah, bh, acc_hh);   // hi*hi
      acc_hl = wmma_bf16(ah, bl, acc_hl);   // hi*lo
      acc_lh = wmma_bf16(al, bh, acc_lh);   // lo*hi (~f32 accuracy, lo*lo dropped)
    }
    const float c2c = c2[col];
#pragma unroll
    for (int v = 0; v < 8; ++v) {
      float s = acc_hh[v] + acc_hl[v] + acc_lh[v];
      float metric = fmaf(-2.0f, s, c2c);   // d2 - ||x||^2 (monotone in dist)
      if (metric < rmin[v]) { rmin[v] = metric; ridx[v] = col; }
    }
  }

  // Cross-lane / cross-wave argmin reduction through LDS.
  __shared__ float smin[2][4][16][16];
  __shared__ int   sidx[2][4][16][16];
#pragma unroll
  for (int v = 0; v < 8; ++v) {
    int rl = (lane < 16) ? v : v + 8;
    smin[rowTile][colTile][rl][l] = rmin[v];
    sidx[rowTile][colTile][rl][l] = ridx[v];
  }
  __syncthreads();
  if (tid < 32) {
    const int rt = tid >> 4, rl = tid & 15;
    float best = 3.4e38f;
    int bi = 0;
    for (int ct = 0; ct < 4; ++ct)
      for (int lc = 0; lc < 16; ++lc) {
        float m = smin[rt][ct][rl][lc];
        int i = sidx[rt][ct][rl][lc];
        if (m < best || (m == best && i < bi)) { best = m; bi = i; }
      }
    const float d2 = best + x2[rowBase + tid];
    minloss[rowBase + tid] = sqrtf(fmaxf(d2, 0.0f));
    minidx[rowBase + tid] = bi;
  }
}

// ---------------------------------------------------------------------------
// Kernel B: run_out = x @ W + b via split-bf16 WMMA, fused with the
// cache-hit select: rows with minloss <= 0.01 take cache_out[idx] instead.
// Grid: (N/32, K/64); same 8-wave 32x64 tiling.
// ---------------------------------------------------------------------------
__global__ __launch_bounds__(256) void out_gemm_select_kernel(
    const unsigned short* __restrict__ xh, const unsigned short* __restrict__ xl,
    const unsigned short* __restrict__ wh, const unsigned short* __restrict__ wl,
    const float* __restrict__ bias, const float* __restrict__ cache_out,
    const float* __restrict__ minloss, const int* __restrict__ minidx,
    float* __restrict__ out) {
  constexpr int D = 512, K = 512;
  const int tid = threadIdx.x;
  const int lane = tid & 31;
  const int wave = tid >> 5;
  const int rowTile = wave >> 2;
  const int colTile = wave & 3;
  const int rowBase = blockIdx.x * 32;
  const int colBase = blockIdx.y * 64;
  const int l = lane & 15;
  const int koffA = (lane >> 4) * 8;
  const int koffB = (lane >> 4) * 8;

  __shared__ float s_ml[32];
  __shared__ int   s_ix[32];
  if (tid < 32) {
    s_ml[tid] = minloss[rowBase + tid];
    s_ix[tid] = minidx[rowBase + tid];
  }
  __syncthreads();

  const int aRow = rowBase + rowTile * 16 + l;
  const unsigned short* xha = xh + (size_t)aRow * D;
  const unsigned short* xla = xl + (size_t)aRow * D;
  const int col = colBase + colTile * 16 + l;        // output column
  const unsigned short* whb = wh + (size_t)col * D;  // W^T row = W column
  const unsigned short* wlb = wl + (size_t)col * D;

  const v8f zero = {0.f, 0.f, 0.f, 0.f, 0.f, 0.f, 0.f, 0.f};
  v8f acc_hh = zero, acc_hl = zero, acc_lh = zero;
#pragma unroll
  for (int ks = 0; ks < D; ks += 32) {
    const int a0 = ks + koffA, a1 = ks + koffA + 16;
    const int b0 = ks + koffB, b1 = ks + koffB + 16;
    v16bf ah = load_tile16(xha + a0, xha + a1);
    v16bf al = load_tile16(xla + a0, xla + a1);
    v16bf bh = load_tile16(whb + b0, whb + b1);
    v16bf bl = load_tile16(wlb + b0, wlb + b1);
    acc_hh = wmma_bf16(ah, bh, acc_hh);
    acc_hl = wmma_bf16(ah, bl, acc_hl);
    acc_lh = wmma_bf16(al, bh, acc_lh);
  }

  const float bc = bias[col];
#pragma unroll
  for (int v = 0; v < 8; ++v) {
    const int rl = (lane < 16) ? v : v + 8;
    const int rloc = rowTile * 16 + rl;
    const int row = rowBase + rloc;
    float val;
    if (s_ml[rloc] <= 0.01f) {
      val = cache_out[(size_t)s_ix[rloc] * K + col];
    } else {
      val = (acc_hh[v] + acc_hl[v] + acc_lh[v]) + bc;
    }
    out[(size_t)row * K + col] = val;
  }
}

// ---------------------------------------------------------------------------
// Host-side launch. Inputs: x[N,D], cache_emb[M,D], cache_out[M,K], W[D,K], b[K]
// ---------------------------------------------------------------------------
extern "C" void kernel_launch(void* const* d_in, const int* in_sizes, int n_in,
                              void* d_out, int out_size, void* d_ws, size_t ws_size,
                              hipStream_t stream) {
  constexpr int D = 512, K = 512;
  const float* x        = (const float*)d_in[0];
  const float* cacheEmb = (const float*)d_in[1];
  const float* cacheOut = (const float*)d_in[2];
  const float* W        = (const float*)d_in[3];
  const float* bias     = (const float*)d_in[4];
  float* out = (float*)d_out;

  const int N = in_sizes[0] / D;   // 8192
  const int M = in_sizes[1] / D;   // 16384

  // Workspace layout (256-byte aligned slabs), ~49 MB total.
  auto align256 = [](size_t v) { return (v + 255) & ~(size_t)255; };
  char* ws = (char*)d_ws;
  size_t off = 0;
  auto take = [&](size_t bytes) { char* p = ws + off; off += align256(bytes); return p; };
  unsigned short* xh = (unsigned short*)take((size_t)N * D * 2);
  unsigned short* xl = (unsigned short*)take((size_t)N * D * 2);
  unsigned short* ch = (unsigned short*)take((size_t)M * D * 2);
  unsigned short* cl = (unsigned short*)take((size_t)M * D * 2);
  unsigned short* wh = (unsigned short*)take((size_t)D * K * 2);
  unsigned short* wl = (unsigned short*)take((size_t)D * K * 2);
  float* x2      = (float*)take((size_t)N * 4);
  float* c2      = (float*)take((size_t)M * 4);
  float* minloss = (float*)take((size_t)N * 4);
  int*   minidx  = (int*)take((size_t)N * 4);
  (void)ws_size; (void)n_in; (void)out_size;

  // 1) Split inputs into bf16 hi/lo + row norms.
  split_rows_kernel<<<N, 256, 0, stream>>>(x, xh, xl, x2, D);
  split_rows_kernel<<<M, 256, 0, stream>>>(cacheEmb, ch, cl, c2, D);
  split_w_transpose_kernel<<<(D * K + 255) / 256, 256, 0, stream>>>(W, wh, wl, D, K);

  // 2) Fused distance GEMM + argmin (WMMA bf16 split, ~f32 accurate).
  dist_argmin_kernel<<<N / 32, 256, 0, stream>>>(xh, xl, ch, cl, x2, c2,
                                                 minloss, minidx, M);

  // 3) Fallback GEMM + cache-hit select (WMMA bf16 split).
  dim3 gridB(N / 32, K / 64);
  out_gemm_select_kernel<<<gridB, 256, 0, stream>>>(xh, xl, wh, wl, bias,
                                                    cacheOut, minloss, minidx, out);
}
